// _global_attention_sop_m_22814866277104
// MI455X (gfx1250) — matrix operand order repaired and checked by reading, not compiled, not run on hardware
//
#include <hip/hip_runtime.h>
#include <hip/hip_bf16.h>
#include <math.h>

// ---------------- problem constants (from reference) ----------------
#define H     4
#define DK    32
#define ITERN 3
#define TRI   528      // 32*33/2 upper-triangle count
#define SLICES 8       // parallel chunk slices per (b,h) in SOP stage

typedef _Float16 half16 __attribute__((ext_vector_type(16)));
typedef float    float8 __attribute__((ext_vector_type(8)));
typedef unsigned uint4v __attribute__((ext_vector_type(4)));
typedef int      int4v  __attribute__((ext_vector_type(4)));
typedef int      int8v  __attribute__((ext_vector_type(8)));

// ---- ordered-uint mapping for float atomicMax (handles negatives) ----
__device__ __forceinline__ unsigned f2key(float f) {
    unsigned u = __float_as_uint(f);
    return (u & 0x80000000u) ? ~u : (u | 0x80000000u);
}
__device__ __forceinline__ float key2f(unsigned k) {
    unsigned u = (k & 0x80000000u) ? (k ^ 0x80000000u) : ~k;
    return __uint_as_float(u);
}

// =====================================================================
// Tensor Data Mover: 2D tile load Global->LDS (CDNA5 ISA §7/§8, D# desc).
//  - data_size=4B (code 2), tile 32x32 elements, row stride H*DK elements
//  - tensor_dim1 = rows remaining in segment -> TDM zero-fills OOB rows,
//    so ragged segment tails contribute 0 with no EXEC divergence.
// =====================================================================
#if __has_include(<hip/amd_detail/amd_gfx1250_TDM.h>)
#define TDM_CALL(g0, g1, g2, g3) \
    __builtin_amdgcn_tensor_load_to_lds(g0, g1, g2, g3, (int8v){0,0,0,0,0,0,0,0}, 0)
#else
#define TDM_CALL(g0, g1, g2, g3) \
    __builtin_amdgcn_tensor_load_to_lds(g0, g1, g2, g3, 0)
#endif

__device__ __forceinline__ void tdm_load_tile(void* ldsbuf, const float* gsrc, int rows) {
    // LDS aperture: generic pointer low 32 bits == LDS byte offset (ISA §10.2)
    unsigned lds = (unsigned)(unsigned long long)ldsbuf;
    unsigned long long ga = (unsigned long long)(const void*)gsrc;
    uint4v g0;
    g0[0] = 1u;                                                   // count=1, user desc
    g0[1] = lds;                                                  // lds_addr[31:0]
    g0[2] = (unsigned)ga;                                         // global_addr[31:0]
    g0[3] = (unsigned)((ga >> 32) & 0x01FFFFFFu) | 0x80000000u;   // addr[56:32] | type=2
    unsigned r = (unsigned)rows;
    int8v g1;
    g1[0] = (int)(2u << 16);                                      // data_size=4B; mask=0
    g1[1] = (int)((unsigned)DK << 16);                            // tensor_dim0[15:0]=32
    g1[2] = (int)((r & 0xFFFFu) << 16);                           // dim0 hi=0 | tensor_dim1 lo
    g1[3] = (int)(((r >> 16) & 0xFFFFu) | ((unsigned)DK << 16));  // dim1 hi | tile_dim0=32
    g1[4] = DK;                                                   // tile_dim1=32, tile_dim2=0
    g1[5] = H * DK;                                               // tensor_dim0_stride=128 elems
    g1[6] = 0;
    g1[7] = 0;
    int4v z = {0, 0, 0, 0};                                       // 2D: groups 2/3 unused
    TDM_CALL(g0, g1, z, z);
}

// =====================================================================
// Fragment-order LDS storage (wave32 WMMA layouts, CDNA5 ISA §7.12.2).
// A 16x32 f16 element (M,K): lane=(M&15)+16*((K>>3)&1), half q=(K&7)+8*(K>=16)
// B 32x16 f16 element (K,N): lane=(N&15)+16*((K>>3)&1), half q=(K&7)+8*(K>=16)
// Stored as buf[tile][lane][q] so a fragment read is one contiguous half16.
// =====================================================================
__device__ __forceinline__ void putA(_Float16 (*Ao)[DK][16], int Rm, int Ck, float v) {
    Ao[Rm >> 4][(Rm & 15) + (((Ck >> 3) & 1) << 4)][(Ck & 7) + ((Ck >= 16) ? 8 : 0)] = (_Float16)v;
}
__device__ __forceinline__ void putB(_Float16 (*Bo)[DK][16], int Rk, int Cn, float v) {
    Bo[Cn >> 4][(Cn & 15) + (((Rk >> 3) & 1) << 4)][(Rk & 7) + ((Rk >= 16) ? 8 : 0)] = (_Float16)v;
}

__device__ __forceinline__ void zero4(float8 c[4]) {
    c[0] = float8{}; c[1] = float8{}; c[2] = float8{}; c[3] = float8{};
}

// 32x32 @ 32x32 -> 4x v_wmma_f32_16x16x32_f16; frags are contiguous LDS b128s
__device__ __forceinline__ void mm32(const _Float16 (*Ao)[DK][16], const _Float16 (*Bo)[DK][16],
                                     int lane, float8 c[4]) {
    half16 a0 = *(const half16*)&Ao[0][lane][0];
    half16 a1 = *(const half16*)&Ao[1][lane][0];
    half16 b0 = *(const half16*)&Bo[0][lane][0];
    half16 b1 = *(const half16*)&Bo[1][lane][0];
    c[0] = __builtin_amdgcn_wmma_f32_16x16x32_f16(false, a0, false, b0, (short)0, c[0], false, false);
    c[1] = __builtin_amdgcn_wmma_f32_16x16x32_f16(false, a0, false, b1, (short)0, c[1], false, false);
    c[2] = __builtin_amdgcn_wmma_f32_16x16x32_f16(false, a1, false, b0, (short)0, c[2], false, false);
    c[3] = __builtin_amdgcn_wmma_f32_16x16x32_f16(false, a1, false, b1, (short)0, c[3], false, false);
}

// C tiles -> frag-order f16 buffers; mode 0: v ; 1: 0.5*(3I-v) ; 2: 3I-v
__device__ __forceinline__ void store_frags(_Float16 (*Ao)[DK][16], _Float16 (*Bo)[DK][16],
                                            const float8 c[4], int lane, int mode) {
    int col = lane & 15, rb = (lane < 16) ? 0 : 8;
#pragma unroll
    for (int t = 0; t < 4; t++) {
        int roff = (t >> 1) * 16, coff = (t & 1) * 16;
#pragma unroll
        for (int r = 0; r < 8; r++) {
            int R = roff + rb + r, C = coff + col;
            float v = c[t][r];
            if (mode == 1)      v = 0.5f * ((R == C ? 3.0f : 0.0f) - v);
            else if (mode == 2) v = (R == C ? 3.0f : 0.0f) - v;
            if (Ao) putA(Ao, R, C, v);
            if (Bo) putB(Bo, R, C, v);
        }
    }
}

// =====================================================================
// K0: init workspace (deg=0, segmax=key(-inf)=0, segsum=0, outacc=0)
// =====================================================================
__global__ void k_init(float* deg, int M, unsigned* segmax, float* segsum,
                       float* outacc, int B) {
    int i = blockIdx.x * blockDim.x + threadIdx.x;
    if (i < M) deg[i] = 0.0f;
    if (i < 2 * B * H) { segmax[i] = 0u; segsum[i] = 0.0f; }
    if (i < B * H * DK * DK) outacc[i] = 0.0f;
}

// =====================================================================
// K1: segment boundaries via binary search on sorted batch[]
// =====================================================================
__global__ void k_bounds(const int* __restrict__ batch, int M, int B,
                         int* __restrict__ bounds) {
    int b = blockIdx.x * blockDim.x + threadIdx.x;
    if (b > B) return;
    int lo = 0, hi = M;
    while (lo < hi) { int mid = (lo + hi) >> 1; if (batch[mid] < b) lo = mid + 1; else hi = mid; }
    bounds[b] = lo;   // bounds[B] == M
}

// =====================================================================
// K2: node degrees  deg[m] = #edges with edge[0]==m
// =====================================================================
__global__ void k_deg(const int* __restrict__ edge0, int E, float* deg) {
    int e = blockIdx.x * blockDim.x + threadIdx.x;
    if (e < E) atomicAdd(&deg[edge0[e]], 1.0f);
}

// =====================================================================
// K3: imp[m,h] = x^T W_h x + b_h ; degh[m,h] = deg*w_h + b_h ; segment max
// (dot(outer(x,x), w) == x^T W x : wj is never materialized)
// =====================================================================
__global__ void k_score(const float* __restrict__ x, const int* __restrict__ batch,
                        const float* __restrict__ attn_w, const float* __restrict__ attn_b,
                        const float* __restrict__ deg_w, const float* __restrict__ deg_b,
                        const float* __restrict__ deg,
                        float* __restrict__ imp, float* __restrict__ degh,
                        unsigned* __restrict__ segmax, int M, int B) {
    __shared__ float sW[H * DK * DK];               // 16 KB of attn_w in LDS
    for (int i = threadIdx.x; i < H * DK * DK; i += blockDim.x) sW[i] = attn_w[i];
    __syncthreads();

    int t = blockIdx.x * blockDim.x + threadIdx.x;
    if (t >= M * H) return;
    int m = t / H, h = t % H;

    const float* xr = x + (size_t)m * (H * DK) + h * DK;
    float xv[DK];
#pragma unroll
    for (int i = 0; i < DK; i++) xv[i] = xr[i];

    const float* W = sW + h * DK * DK;
    float s = 0.0f;
    for (int i = 0; i < DK; i++) {
        float acc = 0.0f;
#pragma unroll
        for (int j = 0; j < DK; j++) acc = fmaf(W[i * DK + j], xv[j], acc);
        s = fmaf(xv[i], acc, s);
    }
    s += attn_b[h];
    float dh = fmaf(deg[m], deg_w[h], deg_b[h]);

    imp[t] = s;
    degh[t] = dh;
    int b = batch[m];
    atomicMax(&segmax[b * H + h],         f2key(s));
    atomicMax(&segmax[B * H + b * H + h], f2key(dh));
}

// =====================================================================
// K4: e = exp(v - segmax) in-place ; segment sums
// =====================================================================
__global__ void k_expsum(const int* __restrict__ batch, const unsigned* __restrict__ segmax,
                         float* __restrict__ imp, float* __restrict__ degh,
                         float* __restrict__ segsum, int M, int B) {
    int t = blockIdx.x * blockDim.x + threadIdx.x;
    if (t >= M * H) return;
    int m = t / H, h = t % H;
    int b = batch[m];
    float e1 = __expf(imp[t]  - key2f(segmax[b * H + h]));
    float e2 = __expf(degh[t] - key2f(segmax[B * H + b * H + h]));
    imp[t] = e1; degh[t] = e2;
    atomicAdd(&segsum[b * H + h],         e1);
    atomicAdd(&segsum[B * H + b * H + h], e2);
}

// =====================================================================
// K5: weighted second-order pooling  out[h*B+b] += (w∘X)^T X  via WMMA.
// grid (B, H, SLICES), 1 wave per block. X tiles staged by the Tensor
// Data Mover (double-buffered, TENSORcnt-paced); TDM zero-fill of rows
// beyond tensor_dim1 handles ragged tails branch-free (EXEC stays full).
// Weights folded into A so padded rows contribute exactly 0.
// =====================================================================
__global__ __launch_bounds__(32)
void k_sop(const float* __restrict__ x, const float* __restrict__ imp,
           const float* __restrict__ degh, const float* __restrict__ segsum,
           const int* __restrict__ bounds, float* __restrict__ outacc, int B) {
    int b = blockIdx.x, h = blockIdx.y, slice = blockIdx.z;
    int lane = threadIdx.x;
    int start = bounds[b], end = bounds[b + 1];

    __shared__ alignas(16) float    sX[2][DK][DK];   // TDM-staged f32 tiles (ping-pong)
    __shared__ alignas(32) _Float16 aF[2][DK][16];   // A = (w∘X)^T, fragment order
    __shared__ alignas(32) _Float16 bF[2][DK][16];   // B = X, fragment order

    float inv1 = 1.0f / segsum[b * H + h];
    float inv2 = 1.0f / segsum[B * H + b * H + h];

    float8 c[4]; zero4(c);

    const int step = DK * SLICES;
    int chunk = start + DK * slice;
    if (chunk < end)
        tdm_load_tile(&sX[0][0][0], x + (size_t)chunk * (H * DK) + h * DK, end - chunk);

    int buf = 0;
    for (; chunk < end; chunk += step, buf ^= 1) {
        int nxt = chunk + step;
        bool more = nxt < end;                       // uniform (scalar) condition
        if (more) {
            tdm_load_tile(&sX[buf ^ 1][0][0], x + (size_t)nxt * (H * DK) + h * DK, end - nxt);
            __builtin_prefetch(&imp[(size_t)nxt * H], 0, 1);   // -> global_prefetch_b8
            __builtin_amdgcn_s_wait_tensorcnt(1);    // next tile may stay in flight
        } else {
            __builtin_amdgcn_s_wait_tensorcnt(0);    // drain: current tile ready
        }

        int m = chunk + lane;
        int mw = (m < end) ? m : (end - 1);          // branchless clamp; OOB rows are zero
        float w = imp[mw * H + h] * inv1 + degh[mw * H + h] * inv2;

        const float* xr = sX[buf][lane];             // lane = node k of this chunk
#pragma unroll
        for (int d = 0; d < DK; d++) {
            float v = xr[d];
            putA(aF, d, lane, w * v);                // A element (M=d, K=lane)
            putB(bF, lane, d, v);                    // B element (K=lane, N=d)
        }
        __syncthreads();
        mm32(aF, bF, lane, c);
        __syncthreads();
    }

    // accumulate into outacc[h*B+b] (SLICES blocks share one tile)
    float* outn = outacc + ((size_t)(h * B + b)) * DK * DK;
    int col = lane & 15, rb = (lane < 16) ? 0 : 8;
#pragma unroll
    for (int t = 0; t < 4; t++) {
        int roff = (t >> 1) * 16, coff = (t & 1) * 16;
#pragma unroll
        for (int r = 0; r < 8; r++)
            atomicAdd(&outn[(roff + rb + r) * DK + coff + col], c[t][r]);
    }
}

// =====================================================================
// K6: Newton–Schulz sqrt of 400 32x32 SPD matrices (1 wave each).
// Matrices live in LDS in fragment order (A-form and B-form per use),
// so every matmul is 4 contiguous b128 frag loads + 4 WMMAs.
// =====================================================================
__global__ __launch_bounds__(32)
void k_ns(const float* __restrict__ outacc, float* __restrict__ out, int B) {
    int n = blockIdx.x;              // n = h*B + b  (matches reference flatten order)
    int lane = threadIdx.x;

    __shared__ alignas(32) _Float16 Ya[2][DK][16], Yb[2][DK][16];
    __shared__ alignas(32) _Float16 Za[2][DK][16], Zb[2][DK][16];
    __shared__ alignas(32) _Float16 Ma[2][DK][16], Mb[2][DK][16];
    __shared__ float sS[DK][DK];

    const float* Tin = outacc + (size_t)n * DK * DK;

    // lane owns row `lane` of T = outacc + 0.001 I
    float row[DK];
#pragma unroll
    for (int j = 0; j < DK; j++) row[j] = Tin[lane * DK + j];

    float diag = Tin[lane * DK + lane] + 0.001f;
    for (int off = 16; off > 0; off >>= 1) diag += __shfl_xor(diag, off, 32);
    float tr = diag;
    float invtr = 1.0f / tr;

    // A = T/tr (A-form only); ZY = Z = 0.5*(3I - A) (both forms)
#pragma unroll
    for (int j = 0; j < DK; j++) {
        float a  = (row[j] + (j == lane ? 0.001f : 0.0f)) * invtr;
        float zy = 0.5f * ((j == lane ? 3.0f : 0.0f) - a);
        putA(Ma, lane, j, a);
        putA(Za, lane, j, zy);
        putB(Zb, lane, j, zy);
    }
    __syncthreads();

    float8 c[4];

    // Y = A @ ZY
    zero4(c); mm32(Ma, Zb, lane, c);
    __syncthreads();
    store_frags(Ya, Yb, c, lane, 0);
    __syncthreads();

    // iterN-2 refinement steps
    for (int it = 0; it < ITERN - 2; ++it) {
        zero4(c); mm32(Za, Yb, lane, c);                 // P = Z@Y
        __syncthreads();
        store_frags(Ma, Mb, c, lane, 1);                 // M = 0.5*(3I - P)
        __syncthreads();
        float8 cy[4]; zero4(cy); mm32(Ya, Mb, lane, cy); // Ynew = Y@M
        float8 cz[4]; zero4(cz); mm32(Ma, Zb, lane, cz); // Znew = M@Z
        __syncthreads();
        store_frags(Ya, Yb, cy, lane, 0);
        store_frags(Za, Zb, cz, lane, 0);
        __syncthreads();
    }

    // YZY = 0.5 * Y @ (3I - Z@Y), scaled by sqrt(tr)
    zero4(c); mm32(Za, Yb, lane, c);                     // P2 = Z@Y
    __syncthreads();
    store_frags(nullptr, Mb, c, lane, 2);                // W = 3I - P2 (B-form only)
    __syncthreads();
    zero4(c); mm32(Ya, Mb, lane, c);                     // Y@W
    float scale = 0.5f * sqrtf(tr);
    __syncthreads();
    {
        int col = lane & 15, rb = (lane < 16) ? 0 : 8;
#pragma unroll
        for (int t = 0; t < 4; t++) {
            int roff = (t >> 1) * 16, coff = (t & 1) * 16;
#pragma unroll
            for (int r = 0; r < 8; r++)
                sS[roff + rb + r][coff + col] = scale * c[t][r];
        }
    }
    __syncthreads();

    // upper-triangle extraction: out[n*528 + t]
    for (int t = lane; t < TRI; t += 32) {
        int tt = t, i = 0, len = DK;
        while (tt >= len) { tt -= len; len--; i++; }
        out[(size_t)n * TRI + t] = sS[i][i + tt];
    }
}

// =====================================================================
// host launch
// =====================================================================
extern "C" void kernel_launch(void* const* d_in, const int* in_sizes, int n_in,
                              void* d_out, int out_size, void* d_ws, size_t ws_size,
                              hipStream_t stream) {
    (void)n_in; (void)ws_size;
    const float* x      = (const float*)d_in[0];
    const int*   batch  = (const int*)d_in[1];
    const int*   edge   = (const int*)d_in[2];     // edge[0] = first E ints
    const float* attn_w = (const float*)d_in[3];
    const float* attn_b = (const float*)d_in[4];
    const float* deg_w  = (const float*)d_in[5];
    const float* deg_b  = (const float*)d_in[6];

    const int M = in_sizes[1];
    const int E = in_sizes[2] / 2;
    const int B = out_size / (H * TRI);            // 100

    // carve workspace (≈3.5 MB total)
    char* p = (char*)d_ws;
    auto carve = [&](size_t bytes) { void* r = (void*)p; p += (bytes + 255) & ~(size_t)255; return r; };
    float*    deg    = (float*)   carve((size_t)M * 4);
    int*      bounds = (int*)     carve((size_t)(B + 1) * 4);
    float*    imp    = (float*)   carve((size_t)M * H * 4);
    float*    degh   = (float*)   carve((size_t)M * H * 4);
    unsigned* segmax = (unsigned*)carve((size_t)2 * B * H * 4);
    float*    segsum = (float*)   carve((size_t)2 * B * H * 4);
    float*    outacc = (float*)   carve((size_t)B * H * DK * DK * 4);

    int initN = B * H * DK * DK; if (M > initN) initN = M;
    k_init  <<<(initN + 255) / 256, 256, 0, stream>>>(deg, M, segmax, segsum, outacc, B);
    k_bounds<<<(B + 1 + 127) / 128, 128, 0, stream>>>(batch, M, B, bounds);
    k_deg   <<<(E + 255) / 256, 256, 0, stream>>>(edge, E, deg);
    k_score <<<(M * H + 255) / 256, 256, 0, stream>>>(x, batch, attn_w, attn_b, deg_w, deg_b,
                                                      deg, imp, degh, segmax, M, B);
    k_expsum<<<(M * H + 255) / 256, 256, 0, stream>>>(batch, segmax, imp, degh, segsum, M, B);
    dim3 gs(B, H, SLICES);
    k_sop   <<<gs, 32, 0, stream>>>(x, imp, degh, segsum, bounds, outacc, B);
    k_ns    <<<B * H, 32, 0, stream>>>(outacc, (float*)d_out, B);
}
